// LinearAttention_47055661695262
// MI455X (gfx1250) — matrix operand order repaired and checked by reading, not compiled, not run on hardware
//
#include <hip/hip_runtime.h>
#include <hip/hip_bf16.h>

// ---------------------------------------------------------------------------
// LinearAttention for MI455X (gfx1250, wave32, WMMA, async LDS staging)
//   qkv GEMM (async double-buffered A tiles) -> softmax(k) -> context = k^T v
//   (K-split over 8 chunks + reduce) -> M_b = context @ w_out_slice^T
//   -> final = q @ M_b + b_out
// Matrix math in v_wmma_f32_16x16x32_f16 (f16 in, f32 accumulate).
// ---------------------------------------------------------------------------

typedef __attribute__((ext_vector_type(16))) _Float16 v16h;
typedef __attribute__((ext_vector_type(8)))  float    v8f;
typedef _Float16 half_t;

#define BATCH 4
#define SEQ   4096
#define DIM   1024
#define HEADS 8
#define DH    64
#define INNER 512               // HEADS*DH
#define MTOT  (BATCH*SEQ)       // 16384
#define KSPLIT 8                // context K-split chunks
#define LDA_PAD 40              // A-tile row pitch in halves (bank-conflict pad)

union FragU { v16h v; uint4 u[2]; };

// A fragment 16(M) x 32(K), f16, from row-major [M, K] memory (ld in halves).
// Lane L<16: rows M=L, K chunks [0..7],[16..23]; L>=16: rows M=L-16, K [8..15],[24..31].
__device__ __forceinline__ v16h load_a16x32(const half_t* __restrict__ tile, int ld, int lane) {
  FragU f;
  const half_t* p = tile + (lane & 15) * ld + ((lane >> 4) << 3);
  f.u[0] = *(const uint4*)(p);
  f.u[1] = *(const uint4*)(p + 16);
  return f.v;
}

// B fragment 32(K) x 16(N), f16, from B^T stored row-major [N, K] (ld in halves).
__device__ __forceinline__ v16h load_bT16x32(const half_t* __restrict__ tileT, int ld, int lane) {
  FragU f;
  const half_t* p = tileT + (lane & 15) * ld + ((lane >> 4) << 3);
  f.u[0] = *(const uint4*)(p);
  f.u[1] = *(const uint4*)(p + 16);
  return f.v;
}

__device__ __forceinline__ v8f wmma_f16(v16h a, v16h b, v8f c) {
  return __builtin_amdgcn_wmma_f32_16x16x32_f16(false, a, false, b, (short)0, c, false, false);
}

// ---------------------------------------------------------------------------
// Kernel 0: convert fp32 inputs to f16 (x, w_qkv, w_out)
// ---------------------------------------------------------------------------
__global__ void convert_kernel(const float* __restrict__ x, const float* __restrict__ wqkv,
                               const float* __restrict__ wout,
                               half_t* __restrict__ xh, half_t* __restrict__ wh,
                               half_t* __restrict__ woh) {
  const int nx  = MTOT * DIM;
  const int nw  = 3 * INNER * DIM;
  const int nwo = DIM * INNER;
  const int total = nx + nw + nwo;
  for (int i = blockIdx.x * blockDim.x + threadIdx.x; i < total;
       i += gridDim.x * blockDim.x) {
    if (i < nx)            xh[i]            = (half_t)x[i];
    else if (i < nx + nw)  wh[i - nx]       = (half_t)wqkv[i - nx];
    else                   woh[i - nx - nw] = (half_t)wout[i - nx - nw];
  }
}

// ---------------------------------------------------------------------------
// Kernel 1: qkv GEMM  [16384 x 1536] = xh [16384 x 1024] @ wh^T (wh [1536,1024])
// Block 256 thr (8 waves), tile 128(M) x 256(N); wave = 64x64 = 4x4 frags.
// A tile (128x32) is double-buffered in LDS via global_load_async_to_lds_b128,
// overlapped with WMMA on the previous tile. B frags loaded direct (L2-fed).
// Epilogue scatters: q -> qh[m,512], k -> kraw[bh,n,64], v -> vT[bh,e,n].
// ---------------------------------------------------------------------------
__global__ void __launch_bounds__(256)
qkv_gemm_kernel(const half_t* __restrict__ xh, const half_t* __restrict__ wh,
                half_t* __restrict__ qh, half_t* __restrict__ kraw,
                half_t* __restrict__ vT) {
  __shared__ half_t smA[2][128 * LDA_PAD];
  const int lane = threadIdx.x & 31;
  const int wave = threadIdx.x >> 5;
  const int t = threadIdx.x;
  const int mblock = blockIdx.y * 128;
  const int wrow = (wave >> 2) * 64;                    // 0 / 64
  const int n0 = blockIdx.x * 256 + (wave & 3) * 64;
  const int K = DIM;

  // Cooperative async A-tile loader: 128 rows x 32 halves; thread t -> row t/2,
  // 16-half chunk (t&1). Two b128 async copies per thread per tile.
  const int arow = t >> 1;
  const int acp  = (t & 1) * 16;
  const half_t* gabase = xh + (size_t)(mblock + arow) * K + acp;

  auto issue_tile = [&](int kbase, int buf) {
    const half_t* g = gabase + kbase;
    unsigned l = (unsigned)(size_t)(&smA[buf][arow * LDA_PAD + acp]);
    asm volatile("global_load_async_to_lds_b128 %0, %1, off"
                 :: "v"(l), "v"(g) : "memory");
    asm volatile("global_load_async_to_lds_b128 %0, %1, off"
                 :: "v"(l + 16u), "v"(g + 8) : "memory");
  };

  v8f acc[4][4] = {};
  issue_tile(0, 0);
  asm volatile("s_wait_asynccnt 0x0" ::: "memory");
  __syncthreads();

  for (int kt = 0; kt < K / 32; ++kt) {
    const int buf = kt & 1;
    if (kt + 1 < K / 32) issue_tile((kt + 1) * 32, buf ^ 1);

    v16h bf[4];
#pragma unroll
    for (int nt = 0; nt < 4; ++nt)
      bf[nt] = load_bT16x32(wh + (size_t)(n0 + nt * 16) * K + kt * 32, K, lane);

#pragma unroll
    for (int mi = 0; mi < 4; ++mi) {
      FragU f;
      const half_t* p =
          &smA[buf][(wrow + mi * 16 + (lane & 15)) * LDA_PAD + ((lane >> 4) << 3)];
      f.u[0] = *(const uint4*)(p);
      f.u[1] = *(const uint4*)(p + 16);
#pragma unroll
      for (int nt = 0; nt < 4; ++nt)
        acc[mi][nt] = wmma_f16(f.v, bf[nt], acc[mi][nt]);
    }
    asm volatile("s_wait_asynccnt 0x0" ::: "memory");
    __syncthreads();
  }

  const int col = lane & 15;
  const int rowoff = (lane >> 4) * 8;
#pragma unroll
  for (int mi = 0; mi < 4; ++mi) {
    const int mbase = mblock + wrow + mi * 16 + rowoff;
    const int b = mbase >> 12;
    const int nbase = mbase & (SEQ - 1);
#pragma unroll
    for (int nt = 0; nt < 4; ++nt) {
      const int o = n0 + nt * 16 + col;
      if (o < INNER) {                                    // q: [m, 512]
#pragma unroll
        for (int r = 0; r < 8; ++r)
          qh[(size_t)(mbase + r) * INNER + o] = (half_t)acc[mi][nt][r];
      } else if (o < 2 * INNER) {                         // k: [bh, n, 64]
        const int w = o - INNER, h = w >> 6, d = w & 63;
        half_t* kp = kraw + ((size_t)(b * HEADS + h) * SEQ + nbase) * DH + d;
#pragma unroll
        for (int r = 0; r < 8; ++r) kp[(size_t)r * DH] = (half_t)acc[mi][nt][r];
      } else {                                            // v^T: [bh, e, n], packed 16B
        const int w = o - 2 * INNER, h = w >> 6, e = w & 63;
        union { uint4 u; half_t hh[8]; } pk;
#pragma unroll
        for (int r = 0; r < 8; ++r) pk.hh[r] = (half_t)acc[mi][nt][r];
        *(uint4*)(vT + ((size_t)(b * HEADS + h) * DH + e) * SEQ + nbase) = pk.u;
      }
    }
  }
}

// ---------------------------------------------------------------------------
// Kernel 2: softmax over d (64) per (bh, n); write transposed kT[bh, d, n].
// ---------------------------------------------------------------------------
__global__ void softmax_k_kernel(const half_t* __restrict__ kraw, half_t* __restrict__ kT) {
  const int lane = threadIdx.x & 31;
  const int row  = blockIdx.x * 8 + (threadIdx.x >> 5);
  const int bh = row >> 12;
  const int n  = row & (SEQ - 1);
  const half_t* src = kraw + (size_t)row * DH;
  float v0 = (float)src[lane * 2];
  float v1 = (float)src[lane * 2 + 1];
  float mx = fmaxf(v0, v1);
#pragma unroll
  for (int off = 16; off; off >>= 1) mx = fmaxf(mx, __shfl_xor(mx, off, 32));
  const float e0 = __expf(v0 - mx);
  const float e1 = __expf(v1 - mx);
  float s = e0 + e1;
#pragma unroll
  for (int off = 16; off; off >>= 1) s += __shfl_xor(s, off, 32);
  const float inv = 1.0f / s;
  half_t* dst = kT + (size_t)bh * DH * SEQ + n;
  dst[(size_t)(lane * 2)     * SEQ] = (half_t)(e0 * inv);
  dst[(size_t)(lane * 2 + 1) * SEQ] = (half_t)(e1 * inv);
}

// ---------------------------------------------------------------------------
// Kernel 3: context partials.  ctx_part[ks][bh][d][e] = sum over n-chunk ks of
// kT[bh,d,n]*vT[bh,e,n].  Grid (32, KSPLIT) = 256 blocks; 8 waves cover the
// 4x4 fragment grid (wave: 1 M-frag x 2 N-frags), K-chunk = 512.
// ---------------------------------------------------------------------------
__global__ void __launch_bounds__(256)
context_gemm_kernel(const half_t* __restrict__ kT, const half_t* __restrict__ vT,
                    float* __restrict__ ctx_part) {
  const int lane = threadIdx.x & 31;
  const int wave = threadIdx.x >> 5;
  const int bh = blockIdx.x;
  const int ks = blockIdx.y;
  const int kchunk = SEQ / KSPLIT;                       // 512
  const half_t* A  = kT + (size_t)bh * DH * SEQ + ks * kchunk;
  const half_t* Bt = vT + (size_t)bh * DH * SEQ + ks * kchunk;
  const int mt  = wave & 3;
  const int ntb = (wave >> 2) * 2;

  v8f acc[2] = {};
  for (int k = 0; k < kchunk; k += 32) {
    v16h a  = load_a16x32(A + (size_t)(mt * 16) * SEQ + k, SEQ, lane);
    v16h b0 = load_bT16x32(Bt + (size_t)((ntb + 0) * 16) * SEQ + k, SEQ, lane);
    v16h b1 = load_bT16x32(Bt + (size_t)((ntb + 1) * 16) * SEQ + k, SEQ, lane);
    acc[0] = wmma_f16(a, b0, acc[0]);
    acc[1] = wmma_f16(a, b1, acc[1]);
  }

  float* C = ctx_part + ((size_t)ks * (BATCH * HEADS) + bh) * DH * DH;
  const int rowoff = (lane >> 4) * 8, col = lane & 15;
#pragma unroll
  for (int j = 0; j < 2; ++j)
#pragma unroll
    for (int r = 0; r < 8; ++r)
      C[(mt * 16 + rowoff + r) * DH + (ntb + j) * 16 + col] = acc[j][r];
}

// Deterministic reduction of the KSPLIT partials -> f16 ctx[bh, d, e]
__global__ void ctx_reduce_kernel(const float* __restrict__ ctx_part,
                                  half_t* __restrict__ ctx) {
  const int bh = blockIdx.x;
  for (int i = threadIdx.x; i < DH * DH; i += blockDim.x) {
    float s = 0.f;
#pragma unroll
    for (int ks = 0; ks < KSPLIT; ++ks)
      s += ctx_part[((size_t)ks * (BATCH * HEADS) + bh) * DH * DH + i];
    ctx[(size_t)bh * DH * DH + i] = (half_t)s;
  }
}

// ---------------------------------------------------------------------------
// Kernel 4: M_b fusion.  M_b[h*64+d, c] = sum_e ctx[bh,d,e] * w_out[c, h*64+e]
// Stored transposed: MT[b][c][hd] (B^T for the final GEMM).
// ---------------------------------------------------------------------------
__global__ void __launch_bounds__(256)
combine_gemm_kernel(const half_t* __restrict__ ctx, const half_t* __restrict__ woh,
                    half_t* __restrict__ MT) {
  const int lane = threadIdx.x & 31;
  const int wave = threadIdx.x >> 5;
  const int bh = blockIdx.x, b = bh >> 3, h = bh & 7;
  const int c0 = blockIdx.y * 128;
  const half_t* A  = ctx + (size_t)bh * DH * DH;        // [64 d][64 e]
  const half_t* Bt = woh + h * DH;                      // row c: woh[c*512+h*64+e]
  const int mt  = wave & 3;
  const int ntb = (wave >> 2) * 4;

  v8f acc[4] = {};
#pragma unroll
  for (int k = 0; k < DH; k += 32) {
    v16h a = load_a16x32(A + mt * 16 * DH + k, DH, lane);
#pragma unroll
    for (int j = 0; j < 4; ++j) {
      v16h bb = load_bT16x32(Bt + (size_t)(c0 + (ntb + j) * 16) * INNER + k, INNER, lane);
      acc[j] = wmma_f16(a, bb, acc[j]);
    }
  }

  half_t* out = MT + (size_t)b * DIM * INNER;
  const int rowoff = (lane >> 4) * 8, col = lane & 15;
#pragma unroll
  for (int j = 0; j < 4; ++j)
#pragma unroll
    for (int r = 0; r < 8; ++r) {
      const int d = mt * 16 + rowoff + r;
      const int c = c0 + (ntb + j) * 16 + col;
      out[(size_t)c * INNER + h * DH + d] = (half_t)acc[j][r];
    }
}

// ---------------------------------------------------------------------------
// Kernel 5: final[m, c] = sum_k qh[m,k] * MT_b[c,k] + b_out[c]   (f32 out)
// Block 256 thr, tile 128(M) x 256(N); wave = 64x64 = 4x4 frags.
// ---------------------------------------------------------------------------
__global__ void __launch_bounds__(256)
final_gemm_kernel(const half_t* __restrict__ qh, const half_t* __restrict__ MT,
                  const float* __restrict__ b_out, float* __restrict__ out) {
  const int lane = threadIdx.x & 31;
  const int wave = threadIdx.x >> 5;
  const int m0 = blockIdx.y * 128 + (wave >> 2) * 64;
  const int c0 = blockIdx.x * 256 + (wave & 3) * 64;
  const int b = m0 >> 12;
  const half_t* Bt = MT + (size_t)b * DIM * INNER;      // [1024 c][512 k]

  v8f acc[4][4] = {};
  for (int k = 0; k < INNER; k += 32) {
    __builtin_prefetch(qh + (size_t)(m0 + (lane & 15)) * INNER + k + 128, 0, 1);
    v16h bf[4];
#pragma unroll
    for (int nt = 0; nt < 4; ++nt)
      bf[nt] = load_bT16x32(Bt + (size_t)(c0 + nt * 16) * INNER + k, INNER, lane);
#pragma unroll
    for (int mi = 0; mi < 4; ++mi) {
      v16h a = load_a16x32(qh + (size_t)(m0 + mi * 16) * INNER + k, INNER, lane);
#pragma unroll
      for (int nt = 0; nt < 4; ++nt)
        acc[mi][nt] = wmma_f16(a, bf[nt], acc[mi][nt]);
    }
  }

  const int rowoff = (lane >> 4) * 8, col = lane & 15;
  float bias[4];
#pragma unroll
  for (int nt = 0; nt < 4; ++nt) bias[nt] = b_out[c0 + nt * 16 + col];
#pragma unroll
  for (int mi = 0; mi < 4; ++mi)
#pragma unroll
    for (int nt = 0; nt < 4; ++nt)
#pragma unroll
      for (int r = 0; r < 8; ++r) {
        const int m = m0 + mi * 16 + rowoff + r;
        const int c = c0 + nt * 16 + col;
        out[(size_t)m * DIM + c] = acc[mi][nt][r] + bias[nt];
      }
}

// ---------------------------------------------------------------------------
// Host side
// ---------------------------------------------------------------------------
extern "C" void kernel_launch(void* const* d_in, const int* in_sizes, int n_in,
                              void* d_out, int out_size, void* d_ws, size_t ws_size,
                              hipStream_t stream) {
  const float* x     = (const float*)d_in[0];
  const float* w_qkv = (const float*)d_in[1];
  const float* w_out = (const float*)d_in[2];
  const float* b_out = (const float*)d_in[3];
  float* out = (float*)d_out;

  char* ws = (char*)d_ws;
  size_t off = 0;
  auto alloc = [&](size_t bytes) -> void* {
    void* p = (void*)(ws + off);
    off += (bytes + 255) & ~(size_t)255;
    return p;
  };
  half_t* xh   = (half_t*)alloc((size_t)MTOT * DIM * 2);
  half_t* wh   = (half_t*)alloc((size_t)3 * INNER * DIM * 2);
  half_t* woh  = (half_t*)alloc((size_t)DIM * INNER * 2);
  half_t* qh   = (half_t*)alloc((size_t)MTOT * INNER * 2);
  half_t* kraw = (half_t*)alloc((size_t)MTOT * INNER * 2);
  half_t* kT   = (half_t*)alloc((size_t)MTOT * INNER * 2);
  half_t* vT   = (half_t*)alloc((size_t)MTOT * INNER * 2);
  half_t* ctx  = (half_t*)alloc((size_t)BATCH * HEADS * DH * DH * 2);
  half_t* MT   = (half_t*)alloc((size_t)BATCH * DIM * INNER * 2);
  float* ctx_part = (float*)alloc((size_t)KSPLIT * BATCH * HEADS * DH * DH * 4);
  (void)ws_size; (void)in_sizes; (void)n_in; (void)out_size;

  convert_kernel<<<4096, 256, 0, stream>>>(x, w_qkv, w_out, xh, wh, woh);

  // qkv GEMM: 16384 x 1536 x 1024 (async LDS A-tiles)
  qkv_gemm_kernel<<<dim3(3 * INNER / 256, MTOT / 128), 256, 0, stream>>>(xh, wh, qh, kraw, vT);

  // softmax over d: 32*4096 rows, 8 rows per block
  softmax_k_kernel<<<(BATCH * HEADS * SEQ) / 8, 256, 0, stream>>>(kraw, kT);

  // context partials (K-split for occupancy) + reduction
  context_gemm_kernel<<<dim3(BATCH * HEADS, KSPLIT), 256, 0, stream>>>(kT, vT, ctx_part);
  ctx_reduce_kernel<<<BATCH * HEADS, 256, 0, stream>>>(ctx_part, ctx);

  // fused context @ w_out slice -> MT
  combine_gemm_kernel<<<dim3(BATCH * HEADS, DIM / 128), 256, 0, stream>>>(ctx, woh, MT);

  // final GEMM: 16384 x 1024 x 512, + bias
  final_gemm_kernel<<<dim3(DIM / 256, MTOT / 128), 256, 0, stream>>>(qh, MT, b_out, out);
}
